// SingleBasisMPS_7490422964934
// MI455X (gfx1250) — matrix operand order repaired and checked by reading, not compiled, not run on hardware
//
#include <hip/hip_runtime.h>

// ---------------------------------------------------------------------------
// MPS chain contraction on MI455X (gfx1250, wave32).
//
//  m0[seq,:]  = left[:, x[seq,0]]            (D=16)
//  m         <- m . bulk[:,:,x[seq,t]]        t = 1..62
//  out[seq]   = dot(m, right[:, x[seq,63]])
//
// One wave handles 16 sequences; each chain step is ONE
// v_wmma_f16_16x16x32_f16 (f32 fallback if the builtin is absent):
//   A[e, K]       = sqrt(2) * bulk[d(K), e, s(K)]      (constant, f16)
//   B[K, seq]     = m[seq, d(K)] * (x[seq,t] == s(K))  (rebuilt per step)
//   D[e, seq]     = m_new[seq,e] * sqrt(2)
//
// K -> (d,s) mapping is chosen so the f16 D layout feeds B with NO repacking:
//   16-bit D: VGPR v = packed rows (M=2v+8h, 2v+1+8h), col N = lane&15
//   B 32x16 : lane-group h supplies K=16h..16h+15 -> branch s = h
//     K = 16s + 2v + p, v<4  ->  d = 2v + 8s + p          (own D words, pa)
//     K = 16s + 2v + p, v>=4 ->  d = 2(v-4) + 8(1-s) + p  (SWAPX16 words, po)
// Per step: 4x ds_swizzle + 8x cndmask (zero the non-selected branch) + WMMA.
//
// sqrt(2) pre-scale of A keeps |m| ~ O(0.1) (unscaled m decays ~2^-31 over
// 62 steps and would underflow f16); final result is multiplied by the exact
// power of two 2^-31.
// ---------------------------------------------------------------------------

typedef __attribute__((ext_vector_type(16))) _Float16 v16h;
typedef __attribute__((ext_vector_type(8)))  _Float16 v8h;
typedef __attribute__((ext_vector_type(8)))  float    v8f;

#define HAVE_WMMA_F16D __has_builtin(__builtin_amdgcn_wmma_f16_16x16x32_f16)

static __device__ __forceinline__ unsigned pk2_f16(float a, float b) {
  union { _Float16 h[2]; unsigned u; } c;
  c.h[0] = (_Float16)a;
  c.h[1] = (_Float16)b;
  return c.u;
}

// Decode K (0..31) -> flat index into bulk[d][e][s] given e, under the
// swizzle-friendly mapping described above.
static __device__ __forceinline__ int bulk_index(int K, int e) {
  const int s = K >> 4;
  const int r = K & 15;
  const int v = r >> 1;
  const int p = r & 1;
  const int d = (v < 4) ? (2 * v + 8 * s + p) : (2 * (v - 4) + 8 * (1 - s) + p);
  return d * 32 + e * 2 + s;   // bulk is (16,16,2) row-major
}

__global__ __launch_bounds__(256) void mps_wmma_kernel(
    const int*   __restrict__ x,      // (N, 64) int32, values in {0,1}
    const float* __restrict__ left,   // (1, 16, 2)
    const float* __restrict__ bulk,   // (16, 16, 2)
    const float* __restrict__ right,  // (16, 1, 2)
    float*       __restrict__ out,    // (N,)
    int N)
{
  const int lane = threadIdx.x & 31;
  const int wave = threadIdx.x >> 5;
  const int tile = blockIdx.x * (blockDim.x >> 5) + wave;   // 16 seqs / tile
  const int e    = lane & 15;     // A row (M=e) / sequence column (N=seq)
  const int h    = lane >> 4;     // wave half == branch s this half supplies

  const int seqBase = tile * 16;
  if (seqBase >= N) return;                 // wave-uniform: EXEC stays all-1s
  int seq = seqBase + e;
  if (seq >= N) seq = N - 1;                // clamp loads only (tail tiles)

  // ---- constant A operand: sqrt(2)-scaled, swizzle-friendly K mapping ----
  // A 16-bit layout: half j holds K = j + 16h (contiguous, matches B).
  const float SQRT2 = 1.41421356237309515f;
  v16h A;
#pragma unroll
  for (int j = 0; j < 16; ++j) {
    const int K = j + 16 * h;
    A[j] = (_Float16)(bulk[bulk_index(K, e)] * SQRT2);
  }

  // ---- pack the 64 selection bits of this lane's sequence ----------------
  const int4* __restrict__ xr4 = (const int4*)(x + (long)seq * 64);
  unsigned lo = 0u, hi = 0u;
#pragma unroll
  for (int q = 0; q < 8; ++q) {
    int4 w = xr4[q];
    lo |= ((unsigned)(w.x & 1)) << (4 * q);
    lo |= ((unsigned)(w.y & 1)) << (4 * q + 1);
    lo |= ((unsigned)(w.z & 1)) << (4 * q + 2);
    lo |= ((unsigned)(w.w & 1)) << (4 * q + 3);
  }
#pragma unroll
  for (int q = 0; q < 8; ++q) {
    int4 w = xr4[8 + q];
    hi |= ((unsigned)(w.x & 1)) << (4 * q);
    hi |= ((unsigned)(w.y & 1)) << (4 * q + 1);
    hi |= ((unsigned)(w.z & 1)) << (4 * q + 2);
    hi |= ((unsigned)(w.w & 1)) << (4 * q + 3);
  }

  const int b0 = (int)(lo & 1u);
  const int bL = (int)((hi >> 31) & 1u);

#if HAVE_WMMA_F16D
  // ======================= f16-accumulator fast path =======================
  // ACC word v = packed (m[seq, 2v+8h], m[seq, 2v+1+8h])  == WMMA f16 D VGPR v
  union { unsigned u[4]; v8h v; _Float16 hh[8]; } ACC;
#pragma unroll
  for (int v = 0; v < 4; ++v)
    ACC.u[v] = pk2_f16(left[(2 * v + 8 * h) * 2 + b0],
                       left[(2 * v + 1 + 8 * h) * 2 + b0]);

#pragma unroll
  for (int t = 1; t <= 62; ++t) {
    const unsigned bit  = (t < 32) ? ((lo >> t) & 1u) : ((hi >> (t - 32)) & 1u);
    const bool     keep = (bit == (unsigned)h);

    union { unsigned u[8]; v16h v; } B;
#pragma unroll
    for (int v = 0; v < 4; ++v) {
      const unsigned pav = ACC.u[v];
      const unsigned pov =
          (unsigned)__builtin_amdgcn_ds_swizzle((int)pav, 0x401F); // SWAPX16
      B.u[v]     = keep ? pav : 0u;
      B.u[v + 4] = keep ? pov : 0u;
    }

    const v8h z = {(_Float16)0.f, (_Float16)0.f, (_Float16)0.f, (_Float16)0.f,
                   (_Float16)0.f, (_Float16)0.f, (_Float16)0.f, (_Float16)0.f};
    ACC.v = __builtin_amdgcn_wmma_f16_16x16x32_f16(
        /*neg_a=*/false, A, /*neg_b=*/false, B.v,
        /*c_mod=*/(short)0, z, /*reuse_a=*/false, /*reuse_b=*/false);
  }

  float p = 0.f;
#pragma unroll
  for (int k = 0; k < 8; ++k)
    p += (float)ACC.hh[k] * right[(k + 8 * h) * 2 + bL];

#else
  // ==================== f32-accumulator fallback path ======================
  float acc[8];
#pragma unroll
  for (int v = 0; v < 8; ++v) acc[v] = left[(v + 8 * h) * 2 + b0];

#pragma unroll
  for (int t = 1; t <= 62; ++t) {
    const unsigned bit  = (t < 32) ? ((lo >> t) & 1u) : ((hi >> (t - 32)) & 1u);
    const bool     keep = (bit == (unsigned)h);

    union { unsigned u[8]; v16h v; } B;
#pragma unroll
    for (int v = 0; v < 4; ++v) {
      const unsigned pav = pk2_f16(acc[2 * v], acc[2 * v + 1]);
      const unsigned pov =
          (unsigned)__builtin_amdgcn_ds_swizzle((int)pav, 0x401F);
      B.u[v]     = keep ? pav : 0u;
      B.u[v + 4] = keep ? pov : 0u;
    }

    const v8f z = {0.f, 0.f, 0.f, 0.f, 0.f, 0.f, 0.f, 0.f};
    v8f c = __builtin_amdgcn_wmma_f32_16x16x32_f16(
        false, A, false, B.v, (short)0, z, false, false);
#pragma unroll
    for (int v = 0; v < 8; ++v) acc[v] = c[v];
  }

  float p = 0.f;
#pragma unroll
  for (int v = 0; v < 8; ++v) p += acc[v] * right[(v + 8 * h) * 2 + bL];
#endif

  // ---- fold the two halves (each summed 8 of 16 e's); undo 2^31 ----------
  union { float f; int i; } q;
  q.f = p;
  q.i = __builtin_amdgcn_ds_swizzle(q.i, 0x401F);          // partner half
  const float total = (p + q.f) * 0x1p-31f;                // exact rescale

  if (h == 0 && (seqBase + e) < N) out[seqBase + e] = total;
}

extern "C" void kernel_launch(void* const* d_in, const int* in_sizes, int n_in,
                              void* d_out, int out_size, void* d_ws, size_t ws_size,
                              hipStream_t stream) {
  const int*   x     = (const int*)  d_in[0];   // (N, 64) int32
  const float* left  = (const float*)d_in[1];   // (1, 16, 2)
  const float* bulk  = (const float*)d_in[2];   // (16, 16, 2)
  const float* right = (const float*)d_in[3];   // (16, 1, 2)
  float*       out   = (float*)d_out;

  const int N     = in_sizes[0] / 64;           // 131072
  const int tiles = (N + 15) / 16;              // 16 sequences per wave
  const int wavesPerBlock = 8;                  // 256 threads = 8 waves
  const int blocks = (tiles + wavesPerBlock - 1) / wavesPerBlock;

  mps_wmma_kernel<<<blocks, wavesPerBlock * 32, 0, stream>>>(
      x, left, bulk, right, out, N);
}
